// Attention2Context_6828998001315
// MI455X (gfx1250) — matrix operand order repaired and checked by reading, not compiled, not run on hardware
//
#include <hip/hip_runtime.h>
#include <stdint.h>

#define DEV __device__ __forceinline__

typedef __attribute__((ext_vector_type(16))) __bf16          bf16x16;
typedef __attribute__((ext_vector_type(8)))  float           f32x8;
typedef __attribute__((ext_vector_type(16))) unsigned short  u16x16;

union FragU {
  u16x16         u;
  bf16x16        b;
  uint4          q[2];
  unsigned short a[16];
};

static constexpr int B_    = 2;
static constexpr int NQ_   = 1024;
static constexpr int NC_   = 1024;   // keys per context
static constexpr int J_    = 2048;   // total keys
static constexpr int QD_   = 1024;
static constexpr int INNER_= 512;    // H*DH
static constexpr int VD_   = 1024;
static constexpr int H_    = 8;
static constexpr int DH_   = 64;
static constexpr int VDH_  = 128;

// ---------- scalar conversions ----------
DEV unsigned short f2bf(float f) {
  union { float f; unsigned u; } c; c.f = f;
  unsigned u = c.u;
  u += 0x7FFFu + ((u >> 16) & 1u);      // round-to-nearest-even
  return (unsigned short)(u >> 16);
}

// ---------- WMMA ----------
DEV f32x8 wmma_bf16(bf16x16 a, bf16x16 b, f32x8 c) {
  // emits v_wmma_f32_16x16x32_bf16
  return __builtin_amdgcn_wmma_f32_16x16x32_bf16(
      /*neg_a=*/false, a, /*neg_b=*/false, b,
      /*c_mod=*/(short)0, c, /*reuse_a=*/false, /*reuse_b=*/false);
}

// A fragment (16M x 32K bf16) from LDS [row][K], K contiguous.
// ISA: lane L -> row = row0 + L%16;
//   elements 0..7  <- K = k0 + (L/16)*8 + 0..7
//   elements 8..15 <- K = k0 + 16 + (L/16)*8 + 0..7
DEV bf16x16 load_frag_a(const unsigned short* s, int ld, int row0, int k0) {
  const int lane = threadIdx.x & 31;
  const int row  = row0 + (lane & 15);
  const int half = lane >> 4;
  const unsigned short* p = s + row * ld + k0 + half * 8;
  FragU r;
  r.q[0] = *reinterpret_cast<const uint4*>(p);
  r.q[1] = *reinterpret_cast<const uint4*>(p + 16);
  return r.b;
}

// B fragment (32K x 16N bf16) from LDS stored N-major [col][K].
// ISA B layout: lane L -> col = L%16;
//   elements 0..7  <- K = k0 + (L/16)*8 + 0..7
//   elements 8..15 <- K = k0 + 16 + (L/16)*8 + 0..7
DEV bf16x16 load_frag_b(const unsigned short* s, int ld, int col0, int k0) {
  const int lane = threadIdx.x & 31;
  const int col  = col0 + (lane & 15);
  const int half = lane >> 4;
  const unsigned short* p = s + col * ld + k0 + half * 8;
  FragU r;
  r.q[0] = *reinterpret_cast<const uint4*>(p);
  r.q[1] = *reinterpret_cast<const uint4*>(p + 16);
  return r.b;
}

// ---------- CDNA5 async global->LDS (ASYNCcnt path) ----------
DEV unsigned lds_addr32(const void* p) {            // low 32 bits of generic ptr == LDS offset
  return (unsigned)(unsigned long long)p;
}
DEV void async_copy_b128(unsigned lds_off, const void* gaddr) {
  asm volatile("global_load_async_to_lds_b128 %0, %1, off"
               :: "v"(lds_off), "v"(gaddr) : "memory");
}
DEV void async_wait_all() {
  asm volatile("s_wait_asynccnt 0x0" ::: "memory");
}

// ---------- f32 -> bf16 convert ----------
__global__ void f32_to_bf16_kernel(const float* __restrict__ in,
                                   unsigned short* __restrict__ out, int n) {
  int i = blockIdx.x * blockDim.x + threadIdx.x;
  if (i < n) out[i] = f2bf(in[i]);
}

// ---------- generic bf16 GEMM + bias, 64x64 tile, BK=64, 8 waves ----------
// out_row = (m / rows_per_batch)*out_batch_stride_rows + out_row_offset + (m % rows_per_batch)
template <bool STORE_BF16>
__global__ __launch_bounds__(256)
void gemm_bias_kernel(const unsigned short* __restrict__ A,   // [M, K] bf16
                      const unsigned short* __restrict__ W,   // [K, N] bf16
                      const float* __restrict__ bias,         // [N]
                      void* __restrict__ outp,
                      int K, int N,
                      int rows_per_batch, int out_batch_stride_rows,
                      int out_row_offset, int out_ld)
{
  constexpr int BS_LD = 72;                      // padded, 144B rows (16B aligned)
  __shared__ __align__(16) unsigned short As[64 * 64];       // [m][k]
  __shared__ __align__(16) unsigned short Bs[64 * BS_LD];    // transposed: [n][k]

  const int m0 = blockIdx.y * 64;
  const int n0 = blockIdx.x * 64;
  const int t    = threadIdx.x;
  const int w    = t >> 5;
  const int lane = t & 31;
  const int half = lane >> 4;
  const int l15  = lane & 15;
  const int mi   = w >> 1;          // 0..3  (16-row subtile)
  const int nb   = (w & 1) * 2;     // subtile cols {nb, nb+1}

  f32x8 acc[2] = {};

  for (int k0 = 0; k0 < K; k0 += 64) {
    // stage A tile 64x64 (row-major, coalesced 16B)
#pragma unroll
    for (int i = 0; i < 2; ++i) {
      const int sid = t + i * 256;             // 512 x 16B segments
      const int row = sid >> 3, seg = sid & 7;
      *reinterpret_cast<uint4*>(&As[row * 64 + seg * 8]) =
          *reinterpret_cast<const uint4*>(&A[(size_t)(m0 + row) * K + k0 + seg * 8]);
    }
    // stage W tile 64x64 transposed into Bs[n][k]
#pragma unroll
    for (int i = 0; i < 2; ++i) {
      const int sid = t + i * 256;
      const int kk = sid >> 3, seg = sid & 7;
      const uint4 wv =
          *reinterpret_cast<const uint4*>(&W[(size_t)(k0 + kk) * N + n0 + seg * 8]);
      const unsigned short* e = reinterpret_cast<const unsigned short*>(&wv);
#pragma unroll
      for (int x = 0; x < 8; ++x) Bs[(seg * 8 + x) * BS_LD + kk] = e[x];
    }
    __syncthreads();

    const bf16x16 a0 = load_frag_a(As, 64, mi * 16, 0);
    const bf16x16 a1 = load_frag_a(As, 64, mi * 16, 32);
#pragma unroll
    for (int j = 0; j < 2; ++j) {
      const bf16x16 b0 = load_frag_b(Bs, BS_LD, (nb + j) * 16, 0);
      const bf16x16 b1 = load_frag_b(Bs, BS_LD, (nb + j) * 16, 32);
      acc[j] = wmma_bf16(a0, b0, acc[j]);
      acc[j] = wmma_bf16(a1, b1, acc[j]);
    }
    __syncthreads();
  }

#pragma unroll
  for (int j = 0; j < 2; ++j) {
#pragma unroll
    for (int r = 0; r < 8; ++r) {
      const int row = m0 + mi * 16 + half * 8 + r;
      const int col = n0 + (nb + j) * 16 + l15;
      const int bb  = row / rows_per_batch;
      const int rr  = row - bb * rows_per_batch;
      const size_t orow = (size_t)bb * out_batch_stride_rows + out_row_offset + rr;
      const float v = acc[j][r] + bias[col];
      if (STORE_BF16) ((unsigned short*)outp)[orow * out_ld + col] = f2bf(v);
      else            ((float*)outp)[orow * out_ld + col] = v;
    }
  }
}

// ---------- flash attention: one block per (batch, head, 64-query tile) ----------
// K chunks are double-buffered in LDS and prefetched with async DMA one chunk
// ahead, hidden behind the current chunk's WMMA + softmax work.
__global__ __launch_bounds__(256)
void attn_kernel(const unsigned short* __restrict__ Q,   // [B*NQ, INNER] bf16
                 const unsigned short* __restrict__ Kb,  // [B*J,  INNER] bf16
                 const unsigned short* __restrict__ Vb,  // [B*J,  VD]    bf16
                 const unsigned char* __restrict__ mask1,// [B,NQ,NC] bool
                 const unsigned char* __restrict__ mask2,// [B,NQ,NC] bool
                 unsigned short* __restrict__ Ob)        // [B*NQ, VD] bf16  (h d) layout
{
  constexpr int SS_LD = 68;  // padded f32 score row
  constexpr int VT_LD = 72;  // padded transposed-V row (144B, 16B aligned)
  constexpr int NCH   = J_ / 64;   // 32 key chunks
  __shared__ __align__(16) unsigned short qs[64 * 64];        // [qrow][dh]
  __shared__ __align__(16) unsigned short ks2[2][64 * 64];    // [key][dh], double buffer
  __shared__ __align__(16) unsigned short vst[128 * VT_LD];   // transposed: [vcol][key]
  __shared__ __align__(16) unsigned short ps[64 * 64];        // [qrow][key]
  __shared__ float Ss[64 * SS_LD];
  __shared__ float mrow[64], lrow[64], arow[64];

  const int qt = blockIdx.x;     // query tile (64 rows)
  const int h  = blockIdx.y;
  const int bb = blockIdx.z;

  const int t    = threadIdx.x;
  const int w    = t >> 5;
  const int lane = t & 31;
  const int half = lane >> 4;
  const int l15  = lane & 15;
  const int mi   = w >> 1;          // 16-row block of the 64-query tile
  const int ns0  = (w & 1) * 2;     // score col blocks {ns0, ns0+1} of 4
  const int nb4  = (w & 1) * 4;     // output col blocks nb4..nb4+3 of 8

  // prologue: async-stage Q tile and K chunk 0
#pragma unroll
  for (int i = 0; i < 2; ++i) {
    const int sid = t + i * 256;            // 512 x 16B segments
    const int row = sid >> 3, seg = sid & 7;
    async_copy_b128(lds_addr32(&qs[row * 64 + seg * 8]),
                    &Q[(size_t)(bb * NQ_ + qt * 64 + row) * INNER_ + h * DH_ + seg * 8]);
    async_copy_b128(lds_addr32(&ks2[0][row * 64 + seg * 8]),
                    &Kb[(size_t)(bb * J_ + row) * INNER_ + h * DH_ + seg * 8]);
  }
  if (t < 64) { mrow[t] = -3.0e38f; lrow[t] = 0.f; }
  async_wait_all();
  __syncthreads();

  const bf16x16 qa0 = load_frag_a(qs, 64, mi * 16, 0);
  const bf16x16 qa1 = load_frag_a(qs, 64, mi * 16, 32);

  f32x8 acc[4] = {};

  for (int ch = 0; ch < NCH; ++ch) {
    const int jbase = ch * 64;
    const unsigned short* kcur = ks2[ch & 1];
    unsigned short*       knxt = ks2[(ch + 1) & 1];

    // prefetch next K chunk (async DMA overlapped with this chunk's compute)
    if (ch + 1 < NCH) {
#pragma unroll
      for (int i = 0; i < 2; ++i) {
        const int sid = t + i * 256;
        const int key = sid >> 3, seg = sid & 7;
        async_copy_b128(
            lds_addr32(&knxt[key * 64 + seg * 8]),
            &Kb[(size_t)(bb * J_ + jbase + 64 + key) * INNER_ + h * DH_ + seg * 8]);
      }
    }

    // S = q @ k^T from the ready buffer (64x64 tile, per-wave 2 subtiles)
#pragma unroll
    for (int ni = 0; ni < 2; ++ni) {
      f32x8 s = {};
      const bf16x16 b0 = load_frag_b(kcur, 64, (ns0 + ni) * 16, 0);
      const bf16x16 b1 = load_frag_b(kcur, 64, (ns0 + ni) * 16, 32);
      s = wmma_bf16(qa0, b0, s);
      s = wmma_bf16(qa1, b1, s);
#pragma unroll
      for (int r = 0; r < 8; ++r)
        Ss[(mi * 16 + half * 8 + r) * SS_LD + (ns0 + ni) * 16 + l15] = s[r];
    }

    // stage V chunk transposed: vst[vcol][key] (coalesced reads, scattered stores)
#pragma unroll
    for (int i = 0; i < 4; ++i) {
      const int sid = t + i * 256;            // 1024 x 16B segments
      const int key = sid >> 4, seg = sid & 15;
      const uint4 vv = *reinterpret_cast<const uint4*>(
          &Vb[(size_t)(bb * J_ + jbase + key) * VD_ + h * VDH_ + seg * 8]);
      const unsigned short* e = reinterpret_cast<const unsigned short*>(&vv);
#pragma unroll
      for (int x = 0; x < 8; ++x) vst[(seg * 8 + x) * VT_LD + key] = e[x];
    }
    __syncthreads();   // Ss + vst visible

    // masked online softmax: one thread per query row
    if (t < 64) {
      const int r = t;
      const int n = qt * 64 + r;
      const unsigned char* mp = (jbase < NC_)
          ? (mask1 + (size_t)(bb * NQ_ + n) * NC_ + jbase)
          : (mask2 + (size_t)(bb * NQ_ + n) * NC_ + (jbase - NC_));
      const float m_old = mrow[r];
      float mc = -3.0e38f;
      for (int j = 0; j < 64; ++j) {
        float s = Ss[r * SS_LD + j] * 0.125f;     // DH^-0.5
        if (!mp[j]) s = -3.0e38f;
        Ss[r * SS_LD + j] = s;
        mc = fmaxf(mc, s);
      }
      const float mn    = fmaxf(m_old, mc);
      const float alpha = __expf(m_old - mn);
      float sum = 0.f;
      for (int j = 0; j < 64; ++j) {
        const float p = __expf(Ss[r * SS_LD + j] - mn);
        ps[r * 64 + j] = f2bf(p);
        sum += p;
      }
      mrow[r] = mn;
      lrow[r] = lrow[r] * alpha + sum;
      arow[r] = alpha;
    }
    __syncthreads();   // ps + arow visible

    // rescale accumulators, then O += P @ V
#pragma unroll
    for (int jj = 0; jj < 4; ++jj)
#pragma unroll
      for (int r = 0; r < 8; ++r)
        acc[jj][r] *= arow[mi * 16 + half * 8 + r];

    const bf16x16 pa0 = load_frag_a(ps, 64, mi * 16, 0);
    const bf16x16 pa1 = load_frag_a(ps, 64, mi * 16, 32);
#pragma unroll
    for (int jj = 0; jj < 4; ++jj) {
      const bf16x16 v0 = load_frag_b(vst, VT_LD, (nb4 + jj) * 16, 0);
      const bf16x16 v1 = load_frag_b(vst, VT_LD, (nb4 + jj) * 16, 32);
      acc[jj] = wmma_bf16(pa0, v0, acc[jj]);
      acc[jj] = wmma_bf16(pa1, v1, acc[jj]);
    }

    async_wait_all();  // own prefetch landed
    __syncthreads();   // everyone's prefetch landed -> knxt ready next iter
  }

  // normalize and write (b, n, h*VDH + d) as bf16
#pragma unroll
  for (int jj = 0; jj < 4; ++jj) {
#pragma unroll
    for (int r = 0; r < 8; ++r) {
      const int row = mi * 16 + half * 8 + r;
      const float o = acc[jj][r] / lrow[row];
      Ob[(size_t)(bb * NQ_ + qt * 64 + row) * VD_ + h * VDH_ + (nb4 + jj) * 16 + l15] =
          f2bf(o);
    }
  }
}

// ---------- host launcher ----------
extern "C" void kernel_launch(void* const* d_in, const int* in_sizes, int n_in,
                              void* d_out, int out_size, void* d_ws, size_t ws_size,
                              hipStream_t stream)
{
  (void)in_sizes; (void)n_in; (void)out_size; (void)ws_size;

  const float* x    = (const float*)d_in[0];
  const float* c1   = (const float*)d_in[1];
  const float* c2   = (const float*)d_in[2];
  const unsigned char* mask1 = (const unsigned char*)d_in[3];
  const unsigned char* mask2 = (const unsigned char*)d_in[4];
  const float* Wq  = (const float*)d_in[5];   const float* bq  = (const float*)d_in[6];
  const float* Wk1 = (const float*)d_in[7];   const float* bk1 = (const float*)d_in[8];
  const float* Wv1 = (const float*)d_in[9];   const float* bv1 = (const float*)d_in[10];
  const float* Wk2 = (const float*)d_in[11];  const float* bk2 = (const float*)d_in[12];
  const float* Wv2 = (const float*)d_in[13];  const float* bv2 = (const float*)d_in[14];
  const float* Wo  = (const float*)d_in[15];  const float* bo  = (const float*)d_in[16];

  // workspace carve-up (bf16 buffers, 256B aligned)
  size_t off = 0;
  auto alloc_bf = [&](size_t elems) {
    unsigned short* p = (unsigned short*)((char*)d_ws + off);
    off += ((elems * sizeof(unsigned short) + 255) & ~(size_t)255);
    return p;
  };
  unsigned short* xb   = alloc_bf((size_t)B_ * NQ_ * QD_);
  unsigned short* c1b  = alloc_bf((size_t)B_ * NC_ * QD_);
  unsigned short* c2b  = alloc_bf((size_t)B_ * NC_ * QD_);
  unsigned short* Wqb  = alloc_bf((size_t)QD_ * INNER_);
  unsigned short* Wk1b = alloc_bf((size_t)QD_ * INNER_);
  unsigned short* Wk2b = alloc_bf((size_t)QD_ * INNER_);
  unsigned short* Wv1b = alloc_bf((size_t)QD_ * VD_);
  unsigned short* Wv2b = alloc_bf((size_t)QD_ * VD_);
  unsigned short* Wob  = alloc_bf((size_t)VD_ * VD_);
  unsigned short* qb   = alloc_bf((size_t)B_ * NQ_ * INNER_);
  unsigned short* kb   = alloc_bf((size_t)B_ * J_  * INNER_);
  unsigned short* vb   = alloc_bf((size_t)B_ * J_  * VD_);
  unsigned short* ab   = alloc_bf((size_t)B_ * NQ_ * VD_);

  auto conv = [&](const float* src, unsigned short* dst, size_t n) {
    f32_to_bf16_kernel<<<dim3((unsigned)((n + 255) / 256)), dim3(256), 0, stream>>>(
        src, dst, (int)n);
  };
  conv(x,   xb,  (size_t)B_ * NQ_ * QD_);
  conv(c1,  c1b, (size_t)B_ * NC_ * QD_);
  conv(c2,  c2b, (size_t)B_ * NC_ * QD_);
  conv(Wq,  Wqb, (size_t)QD_ * INNER_);
  conv(Wk1, Wk1b,(size_t)QD_ * INNER_);
  conv(Wk2, Wk2b,(size_t)QD_ * INNER_);
  conv(Wv1, Wv1b,(size_t)QD_ * VD_);
  conv(Wv2, Wv2b,(size_t)QD_ * VD_);
  conv(Wo,  Wob, (size_t)VD_ * VD_);

  const int Mrows = B_ * NQ_;   // 2048

  // q = x @ Wq + bq                      -> qb [B*NQ, 512]
  gemm_bias_kernel<true><<<dim3(INNER_ / 64, Mrows / 64), dim3(256), 0, stream>>>(
      xb, Wqb, bq, qb, QD_, INNER_, Mrows, 0, 0, INNER_);
  // k[:, :1024] = context  @ Wk1 + bk1   -> kb rows b*2048 + j
  gemm_bias_kernel<true><<<dim3(INNER_ / 64, Mrows / 64), dim3(256), 0, stream>>>(
      c1b, Wk1b, bk1, kb, QD_, INNER_, NC_, J_, 0, INNER_);
  // k[:, 1024:] = context2 @ Wk2 + bk2
  gemm_bias_kernel<true><<<dim3(INNER_ / 64, Mrows / 64), dim3(256), 0, stream>>>(
      c2b, Wk2b, bk2, kb, QD_, INNER_, NC_, J_, NC_, INNER_);
  // v[:, :1024] = context  @ Wv1 + bv1   -> vb [B*J, 1024]
  gemm_bias_kernel<true><<<dim3(VD_ / 64, Mrows / 64), dim3(256), 0, stream>>>(
      c1b, Wv1b, bv1, vb, QD_, VD_, NC_, J_, 0, VD_);
  // v[:, 1024:] = context2 @ Wv2 + bv2
  gemm_bias_kernel<true><<<dim3(VD_ / 64, Mrows / 64), dim3(256), 0, stream>>>(
      c2b, Wv2b, bv2, vb, QD_, VD_, NC_, J_, NC_, VD_);

  // flash attention -> ab [B*NQ, 1024] (h d)
  attn_kernel<<<dim3(NQ_ / 64, H_, B_), dim3(256), 0, stream>>>(
      qb, kb, vb, mask1, mask2, ab);

  // out = ab @ Wo + bo  (f32)
  gemm_bias_kernel<false><<<dim3(VD_ / 64, Mrows / 64), dim3(256), 0, stream>>>(
      ab, Wob, bo, d_out, VD_, VD_, Mrows, 0, 0, VD_);
}
